// MixtureOfExperts_49718541418739
// MI455X (gfx1250) — compile-verified, matrix operand order checked by reading, not run on hardware
//
#include <hip/hip_runtime.h>

typedef unsigned short u16;
typedef unsigned int   u32;
typedef unsigned long long u64;
typedef __attribute__((ext_vector_type(4)))  u32    u32x4;
typedef __attribute__((ext_vector_type(4)))  float  f32x4;
typedef __attribute__((ext_vector_type(16))) __bf16 v16bf;
typedef __attribute__((ext_vector_type(8)))  float  v8f;

__device__ __forceinline__ u16 f32_to_bf16(float f) {
  union { float f; u32 u; } a; a.f = f;
  u32 u = a.u;
  u32 r = u + 0x7FFFu + ((u >> 16) & 1u);   // round-to-nearest-even
  return (u16)(r >> 16);
}
__device__ __forceinline__ float bf16_to_f32(u16 h) {
  union { u32 u; float f; } a; a.u = ((u32)h) << 16;
  return a.f;
}

union Frag32B { u32x4 q[2]; v16bf v; };

__device__ __forceinline__ v16bf ld_frag_lds(const u16* p) {
  Frag32B u;
  u.q[0] = *(const u32x4*)(p);
  u.q[1] = *(const u32x4*)(p + 8);
  return u.v;
}

// 16-byte global -> LDS async copy (CDNA5 GLOBAL_LOAD_ASYNC_TO_LDS_B128,
// tracked by ASYNCcnt). ldsoff = wave-relative LDS byte address (low 32 bits
// of the generic pointer per the LDS aperture mapping).
__device__ __forceinline__ void async_cp16(unsigned ldsoff, const u16* g) {
  asm volatile("global_load_async_to_lds_b128 %0, %1, off"
               :: "v"(ldsoff), "v"((u64)(size_t)g) : "memory");
}
__device__ __forceinline__ void wait_async0() {
  asm volatile("s_wait_asynccnt 0x0" ::: "memory");
}

// ---------------------------------------------------------------------------
// Batched bf16 GEMM: C[e] = act(A[e] (MxK) @ B[e] + bias[e]), Bt stored [N,K].
// Workgroup: 256 thr (8 wave32), tile 128x256, K-step 32.
// Wave tile 64x64 -> 4x4 v_wmma_f32_16x16x32_bf16 per K-step (16 wmma /
// 16 ds_load_b128). Double-buffered LDS fed by async global->LDS copies
// (ASYNCcnt) overlapped with WMMA compute; one barrier per K-step.
// M mult of 128, N mult of 256, K mult of 32 (true for all uses here).
// ---------------------------------------------------------------------------
template <bool RELU, bool OUT_BF16>
__global__ __launch_bounds__(256) void gemm_bf16_wmma(
    const u16* __restrict__ A,      // [M,K] bf16 (batch stride sA elements)
    const u16* __restrict__ Bt,     // [N,K] bf16 (batch stride sB)
    const float* __restrict__ bias, // [N]       (batch stride sBias)
    void* __restrict__ Cout,        // [M,N] bf16 or f32 (batch stride sC)
    int N, int K,
    long long sA, long long sB, long long sBias, long long sC)
{
  __shared__ __align__(16) u16 As[2][128][40];  // +8 pad: keeps 16B align, staggers banks
  __shared__ __align__(16) u16 Bs[2][256][40];

  const int e = blockIdx.z;
  A    += (long long)e * sA;
  Bt   += (long long)e * sB;
  bias += (long long)e * sBias;

  const int tid  = threadIdx.x;
  const int lane = tid & 31;
  const int wv   = tid >> 5;
  const int wm   = wv >> 2;                 // 0..1  (64 rows each)
  const int wn   = wv & 3;                  // 0..3  (64 cols each)
  const long long bm = (long long)blockIdx.y * 128;
  const long long bn = (long long)blockIdx.x * 256;

  // async-copy work split: A tile 128x64B (2 chunks/thread),
  // B tile 256x64B (4 chunks/thread, consecutive lanes cover a row's chunks)
  const int arow = tid >> 1;                // 0..127
  const int acol = (tid & 1) << 4;          // 0 / 16   (u16 units)
  const int brow = tid >> 2;                // 0..63
  const int bcol = (tid & 3) << 3;          // 0,8,16,24 (u16 units)

  const u16* gA = A  + (bm + arow) * (long long)K + acol;
  const u16* gB = Bt + (bn + brow) * (long long)K + bcol;
  const long long rstep = (long long)64 * K;

  auto issue_tile = [&](int kt, int stage) {
    const u16* pa = gA + (long long)kt * 32;
    const unsigned la = (unsigned)(size_t)&As[stage][arow][acol];
    async_cp16(la,       pa);
    async_cp16(la + 16u, pa + 8);
#pragma unroll
    for (int i = 0; i < 4; ++i) {
      const u16* pb = gB + (long long)kt * 32 + (long long)i * rstep;
      const unsigned lb = (unsigned)(size_t)&Bs[stage][brow + 64 * i][bcol];
      async_cp16(lb, pb);
    }
  };

  const int KT = K >> 5;

  issue_tile(0, 0);

  v8f acc[4][4];
  const v8f vzero = {0.f, 0.f, 0.f, 0.f, 0.f, 0.f, 0.f, 0.f};
#pragma unroll
  for (int mt = 0; mt < 4; ++mt)
#pragma unroll
    for (int nt = 0; nt < 4; ++nt) acc[mt][nt] = vzero;

  const int ar   = lane & 15;               // row within 16x16 subtile
  const int krow = (lane >> 4) << 4;        // K half: 0 or 16

  wait_async0();
  __syncthreads();

  for (int kt = 0; kt < KT; ++kt) {
    const int cur = kt & 1;
    if (kt + 1 < KT) issue_tile(kt + 1, cur ^ 1);  // overlap copy with compute

    v16bf afr[4], bfr[4];
#pragma unroll
    for (int mt = 0; mt < 4; ++mt)
      afr[mt] = ld_frag_lds(&As[cur][wm * 64 + mt * 16 + ar][krow]);
#pragma unroll
    for (int nt = 0; nt < 4; ++nt)
      bfr[nt] = ld_frag_lds(&Bs[cur][wn * 64 + nt * 16 + ar][krow]);

#pragma unroll
    for (int mt = 0; mt < 4; ++mt)
#pragma unroll
      for (int nt = 0; nt < 4; ++nt)
        acc[mt][nt] = __builtin_amdgcn_wmma_f32_16x16x32_bf16(
            false, afr[mt], false, bfr[nt], (short)0, acc[mt][nt], false, false);

    wait_async0();
    __syncthreads();
  }

  // epilogue: C/D layout -> VGPR r: (M = r + 8*(lane>=16), N = lane&15)
  const int rlo = (lane >> 4) << 3;         // 0 or 8
  const int col = lane & 15;
#pragma unroll
  for (int nt = 0; nt < 4; ++nt) {
    const long long cg = bn + wn * 64 + nt * 16 + col;
    const float bv = bias[cg];
#pragma unroll
    for (int mt = 0; mt < 4; ++mt) {
      const long long rbase = bm + wm * 64 + mt * 16 + rlo;
#pragma unroll
      for (int r = 0; r < 8; ++r) {
        float v = acc[mt][nt][r] + bv;
        if (RELU) v = fmaxf(v, 0.f);
        const long long idx = (long long)e * sC + (rbase + r) * (long long)N + cg;
        if (OUT_BF16) ((u16*)Cout)[idx]  = f32_to_bf16(v);
        else          ((float*)Cout)[idx] = v;
      }
    }
  }
}

// ---------------------------------------------------------------------------
// f32 [K][N] -> bf16 [N][K] tiled transpose-convert (per batch z)
// ---------------------------------------------------------------------------
__global__ __launch_bounds__(256) void transpose_f32_to_bf16(
    const float* __restrict__ src, u16* __restrict__ dst,
    int K, int N, long long sStride, long long dStride)
{
  __shared__ float tile[32][33];
  const int e = blockIdx.z;
  const float* s = src + (long long)e * sStride;
  u16* d = dst + (long long)e * dStride;
  const int tx = threadIdx.x & 31;
  const int ty = threadIdx.x >> 5;          // 0..7
  const long long bn = (long long)blockIdx.x * 32;   // N
  const long long bk = (long long)blockIdx.y * 32;   // K
#pragma unroll
  for (int i = 0; i < 32; i += 8)
    tile[ty + i][tx] = s[(bk + ty + i) * N + bn + tx];
  __syncthreads();
#pragma unroll
  for (int i = 0; i < 32; i += 8)
    d[(bn + ty + i) * K + bk + tx] = f32_to_bf16(tile[tx][ty + i]);
}

__global__ void convert_f32_to_bf16(const float* __restrict__ s,
                                    u16* __restrict__ d, int n) {
  int i = (blockIdx.x * blockDim.x + threadIdx.x) * 4;
  if (i + 3 < n) {
    f32x4 v = *(const f32x4*)(s + i);
    d[i + 0] = f32_to_bf16(v.x);
    d[i + 1] = f32_to_bf16(v.y);
    d[i + 2] = f32_to_bf16(v.z);
    d[i + 3] = f32_to_bf16(v.w);
  } else {
    for (; i < n; ++i) d[i] = f32_to_bf16(s[i]);
  }
}

// gate logits (G->E=8) + softmax, one thread per batch row
__global__ void gate2_softmax(const u16* __restrict__ g,
                              const float* __restrict__ Wg2,
                              const float* __restrict__ bg2,
                              float* __restrict__ gw, int B, int G) {
  int b = blockIdx.x * blockDim.x + threadIdx.x;
  if (b >= B) return;
  float l[8];
#pragma unroll
  for (int e = 0; e < 8; ++e) l[e] = bg2[e];
  const u16* gr = g + (long long)b * G;
  for (int j = 0; j < G; ++j) {
    float gv = bf16_to_f32(gr[j]);
#pragma unroll
    for (int e = 0; e < 8; ++e) l[e] += gv * Wg2[j * 8 + e];
  }
  float mx = l[0];
#pragma unroll
  for (int e = 1; e < 8; ++e) mx = fmaxf(mx, l[e]);
  float s = 0.f;
#pragma unroll
  for (int e = 0; e < 8; ++e) { l[e] = __expf(l[e] - mx); s += l[e]; }
  float inv = 1.f / s;
#pragma unroll
  for (int e = 0; e < 8; ++e) gw[b * 8 + e] = l[e] * inv;
}

__device__ __forceinline__ float block_reduce_256(float v, float* sm) {
  const int lane = threadIdx.x & 31;
  const int w = threadIdx.x >> 5;
#pragma unroll
  for (int off = 16; off > 0; off >>= 1) v += __shfl_down(v, off, 32);
  if (lane == 0) sm[w] = v;
  __syncthreads();
  float r = 0.f;
#pragma unroll
  for (int i = 0; i < 8; ++i) r += sm[i];
  __syncthreads();
  return r;
}

// Modified Gram-Schmidt over E=8 vectors of D=512, gated combine, final @Wo.
// One 256-thread block per batch row; each thread owns elements t and t+256.
__global__ __launch_bounds__(256) void gram_schmidt_combine(
    const float* __restrict__ eo,   // [E][B][D]
    const float* __restrict__ gw,   // [B][E]
    const float* __restrict__ Wo,   // [D]
    const float* __restrict__ bo,   // [1]
    float* __restrict__ out,        // [B]
    int B, int D)
{
  __shared__ float sm[8];
  const int b = blockIdx.x;
  const int t = threadIdx.x;
  const int d0 = t, d1 = t + 256;
  float v[8][2];
#pragma unroll
  for (int e = 0; e < 8; ++e) {
    const float* p = eo + ((long long)e * B + b) * D;
    v[e][0] = p[d0];
    v[e][1] = p[d1];
  }
  float c0 = 0.f, c1 = 0.f;
#pragma unroll
  for (int i = 0; i < 8; ++i) {
#pragma unroll
    for (int j = 0; j < i; ++j) {
      float num = block_reduce_256(v[i][0] * v[j][0] + v[i][1] * v[j][1], sm);
      float den = block_reduce_256(v[j][0] * v[j][0] + v[j][1] * v[j][1], sm);
      float c = num / (den + 1e-6f);
      v[i][0] -= c * v[j][0];
      v[i][1] -= c * v[j][1];
    }
    float n2 = block_reduce_256(v[i][0] * v[i][0] + v[i][1] * v[i][1], sm);
    float inv = 1.f / fmaxf(sqrtf(n2), 1e-6f);
    v[i][0] *= inv; v[i][1] *= inv;
    float ge = gw[b * 8 + i];
    c0 += ge * v[i][0];
    c1 += ge * v[i][1];
  }
  float fin = block_reduce_256(c0 * Wo[d0] + c1 * Wo[d1], sm);
  if (t == 0) out[b] = fin + bo[0];
}

__global__ void zero_f32(float* __restrict__ p, int n) {
  int i = blockIdx.x * blockDim.x + threadIdx.x;
  if (i < n) p[i] = 0.f;
}

// ---------------------------------------------------------------------------
extern "C" void kernel_launch(void* const* d_in, const int* in_sizes, int n_in,
                              void* d_out, int out_size, void* d_ws, size_t ws_size,
                              hipStream_t stream) {
  (void)in_sizes; (void)n_in; (void)ws_size;
  const float* x   = (const float*)d_in[0];
  const float* W1  = (const float*)d_in[1];
  const float* b1  = (const float*)d_in[2];
  const float* W2  = (const float*)d_in[3];
  const float* b2  = (const float*)d_in[4];
  const float* W3  = (const float*)d_in[5];
  const float* b3  = (const float*)d_in[6];
  const float* Wg1 = (const float*)d_in[7];
  const float* bg1 = (const float*)d_in[8];
  const float* Wg2 = (const float*)d_in[9];
  const float* bg2 = (const float*)d_in[10];
  const float* Wo  = (const float*)d_in[11];
  const float* bo  = (const float*)d_in[12];
  float* out = (float*)d_out;

  const int B = 4096, IN = 1024, H1 = 2048, H2 = 2048, OUTD = 512, E = 8, G = 256;

  char* ws = (char*)d_ws;
  size_t off = 0;
  auto alloc = [&](size_t bytes) -> void* {
    void* p = ws + off;
    off = (off + bytes + 255) & ~(size_t)255;
    return p;
  };
  u16* xb   = (u16*)alloc((size_t)B * IN * 2);
  u16* W1t  = (u16*)alloc((size_t)E * IN * H1 * 2);
  u16* W2t  = (u16*)alloc((size_t)E * H1 * H2 * 2);
  u16* W3t  = (u16*)alloc((size_t)E * H2 * OUTD * 2);
  u16* Wg1t = (u16*)alloc((size_t)IN * G * 2);
  u16* h1   = (u16*)alloc((size_t)E * B * H1 * 2);
  u16* h2   = (u16*)alloc((size_t)E * B * H2 * 2);
  // aliases (stream-ordered, so safe): eo reuses h1 (dead after L2),
  // gate buffers reuse W1t (dead after L1).
  float* eo   = (float*)h1;
  u16*   gbuf = (u16*)W1t;
  float* gwts = (float*)((char*)W1t + (((size_t)B * G * 2 + 255) & ~(size_t)255));

  // 1) convert x to bf16
  {
    int n = B * IN;
    convert_f32_to_bf16<<<dim3((n / 4 + 255) / 256), 256, 0, stream>>>(x, xb, n);
  }
  // 2) transpose-convert weights to bf16 [N][K]
  transpose_f32_to_bf16<<<dim3(H1 / 32, IN / 32, E), 256, 0, stream>>>(
      W1, W1t, IN, H1, (long long)IN * H1, (long long)IN * H1);
  transpose_f32_to_bf16<<<dim3(H2 / 32, H1 / 32, E), 256, 0, stream>>>(
      W2, W2t, H1, H2, (long long)H1 * H2, (long long)H1 * H2);
  transpose_f32_to_bf16<<<dim3(OUTD / 32, H2 / 32, E), 256, 0, stream>>>(
      W3, W3t, H2, OUTD, (long long)H2 * OUTD, (long long)H2 * OUTD);
  transpose_f32_to_bf16<<<dim3(G / 32, IN / 32, 1), 256, 0, stream>>>(
      Wg1, Wg1t, IN, G, (long long)IN * G, (long long)IN * G);

  // 3) Layer 1: h1[e] = relu(x @ W1[e] + b1[e])    (uses W1t -> must precede gate alias)
  gemm_bf16_wmma<true, true><<<dim3(H1 / 256, B / 128, E), 256, 0, stream>>>(
      xb, W1t, b1, h1, H1, IN,
      0LL, (long long)IN * H1, (long long)H1, (long long)B * H1);

  // 4) gate GEMM: g = relu(x @ Wg1 + bg1)          (writes over dead W1t)
  gemm_bf16_wmma<true, true><<<dim3(G / 256, B / 128, 1), 256, 0, stream>>>(
      xb, Wg1t, bg1, gbuf, G, IN, 0LL, 0LL, 0LL, 0LL);

  // 5) gate logits + softmax
  gate2_softmax<<<dim3((B + 255) / 256), 256, 0, stream>>>(gbuf, Wg2, bg2, gwts, B, G);

  // 6) Layer 2: h2[e] = relu(h1[e] @ W2[e] + b2[e])
  gemm_bf16_wmma<true, true><<<dim3(H2 / 256, B / 128, E), 256, 0, stream>>>(
      h1, W2t, b2, h2, H2, H1,
      (long long)B * H1, (long long)H1 * H2, (long long)H2, (long long)B * H2);

  // 7) Layer 3: eo[e] = h2[e] @ W3[e] + b3[e]   (f32 out, overwrites dead h1)
  gemm_bf16_wmma<false, false><<<dim3(OUTD / 256, B / 128, E), 256, 0, stream>>>(
      h2, W3t, b3, eo, OUTD, H2,
      (long long)B * H2, (long long)H2 * OUTD, (long long)OUTD, (long long)B * OUTD);

  // 8) Gram-Schmidt + gated combine + final projection
  gram_schmidt_combine<<<dim3(B), 256, 0, stream>>>(eo, gwts, Wo, bo, out, B, OUTD);

  // 9) tuple's trailing 0.0 scalar (and any tail)
  if (out_size > B) {
    int n = out_size - B;
    zero_f32<<<dim3((n + 255) / 256), 256, 0, stream>>>(out + B, n);
  }
}
